// Critic_7842610283029
// MI455X (gfx1250) — compile-verified, matrix operand order checked by reading, not run on hardware
//
#include <hip/hip_runtime.h>

typedef __attribute__((ext_vector_type(16))) _Float16 v16h;
typedef __attribute__((ext_vector_type(8)))  _Float16 v8h;
typedef __attribute__((ext_vector_type(8)))  float    v8f;
typedef __attribute__((ext_vector_type(4)))  float    f32x4;

#define NB_ROWS   128     // rows per workgroup (4 waves x 32 rows)
#define THREADS   128
#define WROW      264     // padded f16 row stride (256 + 8) -> 528B, 16B aligned, bank-spread
#define XSTR      136     // x_in column-major stride (128 rows + 8 pad) -> 272B, 16B aligned
#define FP_ITERS  49      // fixed-point GEMM iterations (iter 1 is tanh(x_in) analytically)

#define OFF_WFP   0
#define SZ_WFP    (256 * WROW * 2)            // 135168 B : W_fp as f16, row-major padded
#define OFF_Z     (OFF_WFP + SZ_WFP)
#define SZ_Z      (NB_ROWS * WROW * 2)        // 67584 B  : z tile as f16, row-major padded
#define OFF_XIN   (OFF_Z + SZ_Z)
#define SZ_XIN    (256 * XSTR * 2)            // 69632 B  : x_in tile as f16, COLUMN-major
#define OFF_WO    (OFF_XIN + SZ_XIN)
#define SZ_WO     (256 * 4)
#define SMEM_TOTAL (OFF_WO + SZ_WO)           // 273408 B < 320KB/WGP

__device__ __forceinline__ float fast_tanh(float v) {
#if __has_builtin(__builtin_amdgcn_tanhf)
  return __builtin_amdgcn_tanhf(v);           // v_tanh_f32 (gfx1250 TRANS op)
#else
  return tanhf(v);
#endif
}

union H16 { v16h v; f32x4 q[2]; };
union H8  { v8h  h; f32x4 q;    };

__device__ __forceinline__ v16h lds_frag(const _Float16* p) {
  H16 u;
  u.q[0] = *(const f32x4*)p;          // K run 0 (8 halves)
  u.q[1] = *(const f32x4*)(p + 16);   // K run 1 (8 halves, +16 elements)
  return u.v;
}

__device__ __forceinline__ v8f lds_init8(const _Float16* p) {
  H8 u; u.q = *(const f32x4*)p;       // 8 contiguous halves (column-major x_in)
  v8f r;
  #pragma unroll
  for (int i = 0; i < 8; ++i) r[i] = (float)u.h[i];
  return r;
}

__global__ __launch_bounds__(THREADS) void critic_fp_kernel(
    const float* __restrict__ x,  const float* __restrict__ a,
    const float* __restrict__ Wt, const float* __restrict__ bt,
    const float* __restrict__ Wfp,const float* __restrict__ Wo,
    const float* __restrict__ bo, float* __restrict__ out)
{
  extern __shared__ char smem[];
  _Float16* sW  = (_Float16*)(smem + OFF_WFP);
  _Float16* sZ  = (_Float16*)(smem + OFF_Z);
  _Float16* sX  = (_Float16*)(smem + OFF_XIN);   // column-major: sX[col*XSTR + row]
  float*    sWo = (float*)   (smem + OFF_WO);

  const int tid  = threadIdx.x;
  const int wave = tid >> 5;
  const int lane = tid & 31;
  const int ln   = lane & 15;      // row-in-fragment (A) / col (B) / col (C)
  const int hi   = lane >> 4;      // lane-half selector
  const int lm0  = wave * 32;      // this wave's first local row (2 x 16-row tiles)
  const int gm0  = blockIdx.x * NB_ROWS + lm0;

  // ---- stage W_fp (f32 global -> f16 LDS, padded rows) and W_o ----
  for (int i = tid; i < 256 * 64; i += THREADS) {
    int r = i >> 6, c = (i & 63) << 2;
    f32x4 v = *(const f32x4*)(Wfp + r * 256 + c);
    _Float16* d = sW + r * WROW + c;
    d[0] = (_Float16)v[0]; d[1] = (_Float16)v[1];
    d[2] = (_Float16)v[2]; d[3] = (_Float16)v[3];
  }
  for (int i = tid; i < 256; i += THREADS) sWo[i] = Wo[i];

  // ---- prologue: x_in = tanh(concat(x,a) @ Wt^T + bt); z1 = tanh(x_in) ----
  v16h A[2][8];
  #pragma unroll
  for (int mt = 0; mt < 2; ++mt) {
    const int grow = gm0 + mt * 16 + ln;
    #pragma unroll
    for (int kb = 0; kb < 8; ++kb) {
      // f16 A-frag K pattern: lanes hi=0 -> K {b..b+7, b+16..b+23}; hi=1 -> +8
      int k0 = kb * 32 + hi * 8;
      int k1 = k0 + 16;                         // 8-runs never straddle the x|a seam
      const float* p0 = (k0 < 128) ? (x + grow * 128 + k0) : (a + grow * 128 + (k0 - 128));
      const float* p1 = (k1 < 128) ? (x + grow * 128 + k1) : (a + grow * 128 + (k1 - 128));
      f32x4 q0 = *(const f32x4*)p0, q1 = *(const f32x4*)(p0 + 4);
      f32x4 q2 = *(const f32x4*)p1, q3 = *(const f32x4*)(p1 + 4);
      v16h f;
      #pragma unroll
      for (int i = 0; i < 4; ++i) {
        f[i]      = (_Float16)q0[i]; f[4 + i]  = (_Float16)q1[i];
        f[8 + i]  = (_Float16)q2[i]; f[12 + i] = (_Float16)q3[i];
      }
      A[mt][kb] = f;
    }
  }
  for (int jb = 0; jb < 16; ++jb) {
    float bias = bt[jb * 16 + ln];
    v8f acc0, acc1;
    #pragma unroll
    for (int r = 0; r < 8; ++r) { acc0[r] = bias; acc1[r] = bias; }
    #pragma unroll
    for (int kb = 0; kb < 8; ++kb) {
      // B-frag for out = z @ W^T == A-style frag of row-major W (rows = N)
      const float* wr = Wt + (jb * 16 + ln) * 256 + kb * 32 + hi * 8;
      f32x4 q0 = *(const f32x4*)wr,        q1 = *(const f32x4*)(wr + 4);
      f32x4 q2 = *(const f32x4*)(wr + 16), q3 = *(const f32x4*)(wr + 20);
      v16h bf;
      #pragma unroll
      for (int i = 0; i < 4; ++i) {
        bf[i]      = (_Float16)q0[i]; bf[4 + i]  = (_Float16)q1[i];
        bf[8 + i]  = (_Float16)q2[i]; bf[12 + i] = (_Float16)q3[i];
      }
      acc0 = __builtin_amdgcn_wmma_f32_16x16x32_f16(false, A[0][kb], false, bf,
                                                    (short)0, acc0, false, false);
      acc1 = __builtin_amdgcn_wmma_f32_16x16x32_f16(false, A[1][kb], false, bf,
                                                    (short)0, acc1, false, false);
    }
    #pragma unroll
    for (int r = 0; r < 8; ++r) {           // C layout: VGPR r -> row r + 8*hi, col = ln
      int col  = jb * 16 + ln;
      int row0 = lm0 + r + 8 * hi;
      int row1 = row0 + 16;
      float t0 = fast_tanh(acc0[r]);        // x_in elements
      float t1 = fast_tanh(acc1[r]);
      sX[col * XSTR + row0] = (_Float16)t0;               // column-major x_in
      sX[col * XSTR + row1] = (_Float16)t1;
      sZ[row0 * WROW + col] = (_Float16)fast_tanh(t0);    // z after first FP step
      sZ[row1 * WROW + col] = (_Float16)fast_tanh(t1);
    }
  }
  __syncthreads();   // W_fp / W_o staging visible; prologue done

  // ---- 49 fixed-point iterations: z = tanh(z @ Wfp^T + x_in) ----
  // Each wave's 32 z-rows are private; same-wave DS ordering removes all barriers.
  for (int it = 0; it < FP_ITERS; ++it) {
    #pragma unroll
    for (int mt = 0; mt < 2; ++mt)
      #pragma unroll
      for (int kb = 0; kb < 8; ++kb)
        A[mt][kb] = lds_frag(sZ + (lm0 + mt * 16 + ln) * WROW + kb * 32 + hi * 8);

    for (int jb = 0; jb < 16; ++jb) {
      // x_in accumulator init: 8 contiguous halves per tile (column-major sX)
      const _Float16* xi = sX + (jb * 16 + ln) * XSTR + lm0 + 8 * hi;
      v8f acc0 = lds_init8(xi);
      v8f acc1 = lds_init8(xi + 16);
      #pragma unroll
      for (int kb = 0; kb < 8; ++kb) {
        v16h bf = lds_frag(sW + (jb * 16 + ln) * WROW + kb * 32 + hi * 8);
        acc0 = __builtin_amdgcn_wmma_f32_16x16x32_f16(false, A[0][kb], false, bf,
                                                      (short)0, acc0, false, false);
        acc1 = __builtin_amdgcn_wmma_f32_16x16x32_f16(false, A[1][kb], false, bf,
                                                      (short)0, acc1, false, false);
      }
      #pragma unroll
      for (int r = 0; r < 8; ++r) {
        int col  = jb * 16 + ln;
        int row0 = lm0 + r + 8 * hi;
        sZ[row0 * WROW + col]        = (_Float16)fast_tanh(acc0[r]);
        sZ[(row0 + 16) * WROW + col] = (_Float16)fast_tanh(acc1[r]);
      }
    }
  }

  // ---- epilogue: out = z @ Wo^T + bo  (one full 256-wide row dot per lane) ----
  {
    const _Float16* zr = sZ + (lm0 + lane) * WROW;
    float s = 0.f;
    #pragma unroll 8
    for (int k = 0; k < 256; ++k) s += (float)zr[k] * sWo[k];
    out[gm0 + lane] = s + bo[0];
  }
}

extern "C" void kernel_launch(void* const* d_in, const int* in_sizes, int n_in,
                              void* d_out, int out_size, void* d_ws, size_t ws_size,
                              hipStream_t stream) {
  const float* x   = (const float*)d_in[0];
  const float* a   = (const float*)d_in[1];
  const float* W_t = (const float*)d_in[2];
  const float* b_t = (const float*)d_in[3];
  const float* W_fp= (const float*)d_in[4];
  const float* W_o = (const float*)d_in[5];
  const float* b_o = (const float*)d_in[6];
  float* out = (float*)d_out;

  (void)hipFuncSetAttribute((const void*)critic_fp_kernel,
                            hipFuncAttributeMaxDynamicSharedMemorySize, SMEM_TOTAL);
  dim3 grid(32768 / NB_ROWS);   // 256 workgroups, ~1 per WGP
  critic_fp_kernel<<<grid, THREADS, SMEM_TOTAL, stream>>>(
      x, a, W_t, b_t, W_fp, W_o, b_o, out);
}
